// CodeSimilarityDetectionModel_75342316306433
// MI455X (gfx1250) — compile-verified, matrix operand order checked by reading, not run on hardware
//
#include <hip/hip_runtime.h>
#include <hip/hip_bf16.h>

// Problem constants (match reference)
#define BATCH     64
#define NNODE     1024
#define HDIM      128
#define DEG       16
#define NCLS      4
#define NUMN      (BATCH * NNODE)          // 65536 nodes
#define EDGES     (NUMN * DEG)             // 1048576 edges
#define NM        ((size_t)NUMN * HDIM)    // 8388608 elements per feature buffer

typedef __attribute__((ext_vector_type(16))) __bf16 v16bf;
typedef __attribute__((ext_vector_type(8)))  float  v8f;

union BF16x16 {
    v16bf v;
    uint4 q[2];
};

__device__ __forceinline__ unsigned short f2bf(float f) {
    unsigned int u = __float_as_uint(f);
    u += 0x7FFFu + ((u >> 16) & 1u);          // round-to-nearest-even
    return (unsigned short)(u >> 16);
}

__device__ __forceinline__ float bf2f(unsigned short u) {
    return __uint_as_float(((unsigned int)u) << 16);
}

// Load 16 contiguous bf16 (32B-aligned) straight into a WMMA operand — no VALU conversion.
__device__ __forceinline__ v16bf load16bf(const unsigned short* __restrict__ p) {
    const uint4* q = (const uint4*)p;
    BF16x16 r;
    r.q[0] = q[0];
    r.q[1] = q[1];
    return r.v;
}

// ---------------- setup kernels ----------------

// dinv regions (2*65536, init 1.0 for self loops) and graph-mean region (2*8192, init 0)
__global__ void init_kernel(float* __restrict__ dinv_both, float* __restrict__ g_both) {
    int i = blockIdx.x * 256 + threadIdx.x;
    if (i < 2 * NUMN) dinv_both[i] = 1.0f;
    if (i < 2 * BATCH * HDIM) g_both[i] = 0.0f;
}

__global__ void deg_kernel(const long long* __restrict__ ei, float* __restrict__ deg) {
    size_t e = (size_t)blockIdx.x * 256 + threadIdx.x;
    if (e < EDGES) atomicAdd(&deg[(size_t)ei[EDGES + e]], 1.0f);
}

__global__ void dinv_kernel(float* __restrict__ d) {
    int i = blockIdx.x * 256 + threadIdx.x;
    if (i < 2 * NUMN) d[i] = rsqrtf(d[i]);
}

// Transpose + convert W [k][c] f32 -> Wt [c][k] bf16 (done once per weight matrix)
__global__ void prep_w_kernel(const float* __restrict__ W, unsigned short* __restrict__ Wt) {
    int idx = blockIdx.x * 256 + threadIdx.x;     // 16384
    int k = idx >> 7, c = idx & 127;
    Wt[c * HDIM + k] = f2bf(W[idx]);
}

// f32 -> packed bf16 (2 elems/thread)
__global__ void cvt_bf16_kernel(const float* __restrict__ in, unsigned int* __restrict__ outbf) {
    size_t i = ((size_t)blockIdx.x * 256 + threadIdx.x) * 2;
    float2 v = *(const float2*)(in + i);
    outbf[i >> 1] = (unsigned int)f2bf(v.x) | ((unsigned int)f2bf(v.y) << 16);
}

// ---------------- GCN layer kernels ----------------

// XW = X @ W  (X bf16 [65536,128], Wt bf16 col-major [128,128]) via WMMA, f32 out.
// Block = 256 threads (8 waves); block covers 128 rows; wave = 16 rows x 128 cols.
// Software-pipelined: 8 accumulators, 3 live B buffers = prefetch distance 2.
__global__ void gemm_xw_kernel(const unsigned short* __restrict__ Xbf,
                               const unsigned short* __restrict__ Wt,
                               float* __restrict__ XW) {
    int tid = threadIdx.x;
    int wave = tid >> 5, lane = tid & 31;
    int l16 = lane & 15, lhi = lane >> 4;      // lane group 0 / 1
    size_t rbase = (size_t)blockIdx.x * 128 + wave * 16;
    const unsigned short* Abase = Xbf + (rbase + l16) * HDIM + lhi * 16;
    const unsigned short* Bbase = Wt + (size_t)l16 * HDIM + lhi * 16;

    v16bf a[4];
    #pragma unroll
    for (int s = 0; s < 4; ++s) a[s] = load16bf(Abase + s * 32);

    v8f acc[8];
    #pragma unroll
    for (int t = 0; t < 8; ++t) acc[t] = (v8f){0,0,0,0,0,0,0,0};

    // B tile offset for idx = s*8 + t  (tile t -> +t*16 cols, step s -> +s*32 K elems)
    #define GB_OFF(ix) ((((ix) & 7) * (16 * HDIM)) + (((ix) >> 3) * 32))
    v16bf b0 = load16bf(Bbase + GB_OFF(0));
    v16bf b1 = load16bf(Bbase + GB_OFF(1));
    #pragma unroll
    for (int idx = 0; idx < 32; ++idx) {
        int s = idx >> 3, t = idx & 7;
        v16bf bn = b0;
        if (idx + 2 < 32) bn = load16bf(Bbase + GB_OFF(idx + 2));
        acc[t] = __builtin_amdgcn_wmma_f32_16x16x32_bf16(
            false, a[s], false, b0, (short)0, acc[t], false, false);
        b0 = b1;
        b1 = bn;
    }
    #undef GB_OFF

    #pragma unroll
    for (int t = 0; t < 8; ++t) {
        int col = t * 16 + l16;
        #pragma unroll
        for (int j = 0; j < 8; ++j)            // C layout: row = j + 8*lhi
            XW[(rbase + (size_t)(lhi * 8 + j)) * HDIM + col] = acc[t][j];
    }
}

// out[n,h] = xw[n,h] * dinv[n]^2   (self-loop term; fully overwrites out)
__global__ void selfloop_kernel(const float* __restrict__ xw, const float* __restrict__ dinv,
                                float* __restrict__ out) {
    size_t i = (size_t)blockIdx.x * 256 + threadIdx.x;
    size_t n = i >> 7;
    float d = dinv[n];
    out[i] = xw[i] * d * d;
}

// Edge scatter: out[dst] += xw[src] * dinv[src]*dinv[dst]; one thread per (edge, 4 floats)
__global__ void scatter_kernel(const float* __restrict__ xw, const long long* __restrict__ ei,
                               const float* __restrict__ dinv, float* __restrict__ out) {
    size_t tid = (size_t)blockIdx.x * 256 + threadIdx.x;   // EDGES*32 threads
    size_t e = tid >> 5;
    int h = (int)(tid & 31) << 2;
    size_t src = (size_t)ei[e];
    size_t dst = (size_t)ei[EDGES + e];
    float coef = dinv[src] * dinv[dst];
    float4 v = *(const float4*)(xw + src * HDIM + h);
    float* o = out + dst * HDIM + h;
    atomicAdd(o + 0, v.x * coef);
    atomicAdd(o + 1, v.y * coef);
    atomicAdd(o + 2, v.z * coef);
    atomicAdd(o + 3, v.w * coef);
}

// (agg + bias [+ relu]) -> packed bf16 output (next layer's GEMM input / final H)
__global__ void bias_relu_tobf_kernel(const float* __restrict__ in, const float* __restrict__ bias,
                                      unsigned int* __restrict__ outbf, int do_relu) {
    size_t i = ((size_t)blockIdx.x * 256 + threadIdx.x) * 2;
    float2 v = *(const float2*)(in + i);
    int h = (int)(i & (HDIM - 1));
    float a = v.x + bias[h];
    float b = v.y + bias[h + 1];
    if (do_relu) { a = fmaxf(a, 0.0f); b = fmaxf(b, 0.0f); }
    outbf[i >> 1] = (unsigned int)f2bf(a) | ((unsigned int)f2bf(b) << 16);
}

// ---------------- head kernels ----------------

// Partial column sums into g[b*128+h] (scaled by 1/N). grid = (64, 8), block = 128.
__global__ void mean_kernel(const unsigned short* __restrict__ Hm, float* __restrict__ g) {
    int b = blockIdx.x, chunk = blockIdx.y, h = threadIdx.x;
    float s = 0.0f;
    size_t base = ((size_t)b * NNODE + (size_t)chunk * 128) * HDIM + h;
    for (int n = 0; n < 128; ++n) s += bf2f(Hm[base + (size_t)n * HDIM]);
    atomicAdd(&g[b * HDIM + h], s * (1.0f / (float)NNODE));
}

__global__ void logits_kernel(const float* __restrict__ gs, const float* __restrict__ gt,
                              const float* __restrict__ Wc1, const float* __restrict__ bc1,
                              const float* __restrict__ Wc2, const float* __restrict__ bc2,
                              float* __restrict__ out) {
    __shared__ float g[HDIM];
    __shared__ float hid[HDIM];
    int b = blockIdx.x, t = threadIdx.x;
    g[t] = fabsf(gs[b * HDIM + t] - gt[b * HDIM + t]);
    __syncthreads();
    float acc = bc1[t];
    for (int k = 0; k < HDIM; ++k) acc += g[k] * Wc1[k * HDIM + t];
    hid[t] = fmaxf(acc, 0.0f);
    __syncthreads();
    if (t < NCLS) {
        float o = bc2[t];
        for (int k = 0; k < HDIM; ++k) o += hid[k] * Wc2[k * NCLS + t];
        out[b * NCLS + t] = o;
    }
}

// ---------------- fused S0 = softmax(Hs . Ht^T) ----------------
// grid = BATCH * (NNODE/16) blocks, 256 threads (8 waves). Each block: 16 rows x 1024 cols
// in LDS, then row softmax, then coalesced stream-out. Same pipelined WMMA structure;
// wave handles col tiles ct = wave + 8*t (t = 0..7), B strip is L2-resident (256 KB/batch).
__global__ void s0_kernel(const unsigned short* __restrict__ Hs,
                          const unsigned short* __restrict__ Ht,
                          float* __restrict__ out) {
    extern __shared__ float smem[];           // 16*1024 scores + 16 max + 16 inv-sum
    float* scores = smem;
    float* rowm = smem + 16 * 1024;
    float* rinv = rowm + 16;

    int tid = threadIdx.x;
    int wave = tid >> 5, lane = tid & 31;
    int l16 = lane & 15, lhi = lane >> 4;
    int b  = blockIdx.x >> 6;
    int rb = blockIdx.x & 63;

    const unsigned short* Abase =
        Hs + ((size_t)b * NNODE + rb * 16 + l16) * HDIM + lhi * 16;
    const unsigned short* Bbase =
        Ht + ((size_t)b * NNODE + wave * 16 + l16) * HDIM + lhi * 16;

    v16bf a[4];
    #pragma unroll
    for (int s = 0; s < 4; ++s) a[s] = load16bf(Abase + s * 32);

    v8f acc[8];
    #pragma unroll
    for (int t = 0; t < 8; ++t) acc[t] = (v8f){0,0,0,0,0,0,0,0};

    // B tile offset for idx = s*8 + t  (tile t -> +t*128 rows, step s -> +s*32 K elems)
    #define SB_OFF(ix) ((((ix) & 7) * (128 * HDIM)) + (((ix) >> 3) * 32))
    v16bf b0 = load16bf(Bbase + SB_OFF(0));
    v16bf b1 = load16bf(Bbase + SB_OFF(1));
    #pragma unroll
    for (int idx = 0; idx < 32; ++idx) {
        int s = idx >> 3, t = idx & 7;
        v16bf bn = b0;
        if (idx + 2 < 32) bn = load16bf(Bbase + SB_OFF(idx + 2));
        acc[t] = __builtin_amdgcn_wmma_f32_16x16x32_bf16(
            false, a[s], false, b0, (short)0, acc[t], false, false);
        b0 = b1;
        b1 = bn;
    }
    #undef SB_OFF

    #pragma unroll
    for (int t = 0; t < 8; ++t) {
        int col = (wave + 8 * t) * 16 + l16;
        #pragma unroll
        for (int j = 0; j < 8; ++j)
            scores[(lhi * 8 + j) * 1024 + col] = acc[t][j];
    }
    __syncthreads();

    // row stats: 16 threads per row, shuffle-reduce within 16-lane groups (wave32)
    int row = tid >> 4, lg = tid & 15;
    float m = -3.4e38f;
    for (int i = lg; i < 1024; i += 16) m = fmaxf(m, scores[row * 1024 + i]);
    #pragma unroll
    for (int o = 8; o >= 1; o >>= 1) m = fmaxf(m, __shfl_xor(m, o, 16));
    float sum = 0.0f;
    for (int i = lg; i < 1024; i += 16) sum += __expf(scores[row * 1024 + i] - m);
    #pragma unroll
    for (int o = 8; o >= 1; o >>= 1) sum += __shfl_xor(sum, o, 16);
    if (lg == 0) { rowm[row] = m; rinv[row] = 1.0f / sum; }
    __syncthreads();

    size_t obase = ((size_t)b * NNODE + (size_t)rb * 16) * (size_t)NNODE;
    for (int i = tid; i < 16 * 1024; i += 256) {
        int r = i >> 10;
        out[obase + i] = __expf(scores[i] - rowm[r]) * rinv[r];
    }
}

// ---------------- driver ----------------

static void run_side(const float* x, const long long* ei, const float* dinv,
                     const unsigned short* Wt1, const float* b1,
                     const unsigned short* Wt2, const float* b2,
                     const unsigned short* Wt3, const float* b3,
                     float* T1, float* XW, unsigned short* Xbf, unsigned short* Houtbf,
                     hipStream_t stream) {
    const int gElem = (int)(NM / 256);           // 32768 blocks for 8.4M elementwise
    const int gHalf = (int)(NM / 512);           // 2 elems/thread kernels
    const int gScat = (EDGES * 32) / 256;        // 131072 blocks

    cvt_bf16_kernel<<<gHalf, 256, 0, stream>>>(x, (unsigned int*)Xbf);
    // layer 1 (relu)
    gemm_xw_kernel<<<NUMN / 128, 256, 0, stream>>>(Xbf, Wt1, XW);
    selfloop_kernel<<<gElem, 256, 0, stream>>>(XW, dinv, T1);
    scatter_kernel<<<gScat, 256, 0, stream>>>(XW, ei, dinv, T1);
    bias_relu_tobf_kernel<<<gHalf, 256, 0, stream>>>(T1, b1, (unsigned int*)Xbf, 1);
    // layer 2 (relu)
    gemm_xw_kernel<<<NUMN / 128, 256, 0, stream>>>(Xbf, Wt2, XW);
    selfloop_kernel<<<gElem, 256, 0, stream>>>(XW, dinv, T1);
    scatter_kernel<<<gScat, 256, 0, stream>>>(XW, ei, dinv, T1);
    bias_relu_tobf_kernel<<<gHalf, 256, 0, stream>>>(T1, b2, (unsigned int*)Xbf, 1);
    // layer 3 (no relu) -> final H in bf16
    gemm_xw_kernel<<<NUMN / 128, 256, 0, stream>>>(Xbf, Wt3, XW);
    selfloop_kernel<<<gElem, 256, 0, stream>>>(XW, dinv, T1);
    scatter_kernel<<<gScat, 256, 0, stream>>>(XW, ei, dinv, T1);
    bias_relu_tobf_kernel<<<gHalf, 256, 0, stream>>>(T1, b3, (unsigned int*)Houtbf, 0);
}

extern "C" void kernel_launch(void* const* d_in, const int* in_sizes, int n_in,
                              void* d_out, int out_size, void* d_ws, size_t ws_size,
                              hipStream_t stream) {
    const float*     x_s  = (const float*)d_in[0];
    const long long* ei_s = (const long long*)d_in[1];
    const float*     x_t  = (const float*)d_in[2];
    const long long* ei_t = (const long long*)d_in[3];
    const float* W1 = (const float*)d_in[4];  const float* b1 = (const float*)d_in[5];
    const float* W2 = (const float*)d_in[6];  const float* b2 = (const float*)d_in[7];
    const float* W3 = (const float*)d_in[8];  const float* b3 = (const float*)d_in[9];
    const float* Wc1 = (const float*)d_in[10]; const float* bc1 = (const float*)d_in[11];
    const float* Wc2 = (const float*)d_in[12]; const float* bc2 = (const float*)d_in[13];
    float* out = (float*)d_out;

    // workspace carve-up (~118 MB)
    float* ws     = (float*)d_ws;
    float* dinv_s = ws;                        // 65536 f32
    float* dinv_t = dinv_s + NUMN;             // 65536 f32
    float* gs     = dinv_t + NUMN;             // 8192 f32
    float* gt     = gs + BATCH * HDIM;         // 8192 f32
    float* T1     = gt + BATCH * HDIM;         // 8.4M f32
    float* XW     = T1 + NM;                   // 8.4M f32
    unsigned short* Xbf  = (unsigned short*)(XW + NM);   // 8.4M bf16
    unsigned short* Hsbf = Xbf + NM;                     // 8.4M bf16
    unsigned short* Htbf = Hsbf + NM;                    // 8.4M bf16
    unsigned short* Wt1  = Htbf + NM;                    // 16384 bf16
    unsigned short* Wt2  = Wt1 + HDIM * HDIM;
    unsigned short* Wt3  = Wt2 + HDIM * HDIM;

    // degrees -> dinv (self-loops included via init to 1.0); zero graph-mean accumulators
    init_kernel<<<512, 256, 0, stream>>>(dinv_s, gs);
    deg_kernel<<<EDGES / 256, 256, 0, stream>>>(ei_s, dinv_s);
    deg_kernel<<<EDGES / 256, 256, 0, stream>>>(ei_t, dinv_t);
    dinv_kernel<<<512, 256, 0, stream>>>(dinv_s);   // covers both (contiguous)

    // weights: transpose + convert once (shared by both sides)
    prep_w_kernel<<<64, 256, 0, stream>>>(W1, Wt1);
    prep_w_kernel<<<64, 256, 0, stream>>>(W2, Wt2);
    prep_w_kernel<<<64, 256, 0, stream>>>(W3, Wt3);

    run_side(x_s, ei_s, dinv_s, Wt1, b1, Wt2, b2, Wt3, b3, T1, XW, Xbf, Hsbf, stream);
    run_side(x_t, ei_t, dinv_t, Wt1, b1, Wt2, b2, Wt3, b3, T1, XW, Xbf, Htbf, stream);

    mean_kernel<<<dim3(BATCH, 8), 128, 0, stream>>>(Hsbf, gs);
    mean_kernel<<<dim3(BATCH, 8), 128, 0, stream>>>(Htbf, gt);
    logits_kernel<<<BATCH, 128, 0, stream>>>(gs, gt, Wc1, bc1, Wc2, bc2, out);

    // fused batched GEMM + row softmax, streamed to d_out after the 256 logit floats
    size_t shmem = (16 * 1024 + 32) * sizeof(float);
    s0_kernel<<<BATCH * (NNODE / 16), 256, shmem, stream>>>(Hsbf, Htbf, out + BATCH * NCLS);
}
